// Backbone_32942399160850
// MI455X (gfx1250) — matrix operand order repaired and checked by reading, not compiled, not run on hardware
//
#include <hip/hip_runtime.h>

typedef __attribute__((ext_vector_type(16))) _Float16 v16h;
typedef __attribute__((ext_vector_type(8)))  _Float16 v8h;
typedef __attribute__((ext_vector_type(8)))  float    v8f;

#define KNNK 20

// ---------------------------------------------------------------------------
// Scalar fused distance + top-20 (C=3 input layer only).
// ---------------------------------------------------------------------------
template<int C>
__global__ void knn_topk_kernel(const float* __restrict__ f, int N, int* __restrict__ idxOut) {
  const int b = blockIdx.y;
  const int n = blockIdx.x * 256 + threadIdx.x;
  const float* fb = f + (size_t)b * C * N;

  float fr[C];
  float xxn = 0.f;
#pragma unroll
  for (int c = 0; c < C; ++c) { float v = fb[(size_t)c * N + n]; fr[c] = v; xxn += v * v; }

  float val[KNNK]; int ind[KNNK];
#pragma unroll
  for (int k = 0; k < KNNK; ++k) { val[k] = -3.4e38f; ind[k] = 0; }

  __shared__ float ldsF[C * 128];
  __shared__ float ldsXX[128];

  for (int m0 = 0; m0 < N; m0 += 128) {
    __syncthreads();
    for (int t = threadIdx.x; t < C * 128; t += 256) {
      int c = t >> 7, j = t & 127;
      ldsF[t] = fb[(size_t)c * N + m0 + j];
    }
    __syncthreads();
    if (threadIdx.x < 128) {
      float s = 0.f;
#pragma unroll
      for (int c = 0; c < C; ++c) { float v = ldsF[c * 128 + threadIdx.x]; s += v * v; }
      ldsXX[threadIdx.x] = s;
    }
    __syncthreads();
    for (int j = 0; j < 128; ++j) {
      float inner = 0.f;
#pragma unroll
      for (int c = 0; c < C; ++c) inner += fr[c] * ldsF[c * 128 + j];
      float pd = 2.f * inner - xxn - ldsXX[j];
      if (pd > val[KNNK - 1]) {
        float v = pd; int id = m0 + j;
#pragma unroll
        for (int k = 0; k < KNNK; ++k) {
          if (v > val[k]) { float tv = val[k]; int ti = ind[k]; val[k] = v; ind[k] = id; v = tv; id = ti; }
        }
      }
    }
  }
#pragma unroll
  for (int k = 0; k < KNNK; ++k)
    idxOut[((size_t)b * N + n) * KNNK + k] = ind[k];
}

// ---------------------------------------------------------------------------
// WMMA kNN (C=64).  Both row block (128x64) and column chunk (64x64) staged
// in LDS operand-major (stride 72 halves: 144B rows keep ds_load_b128
// alignment).  A is rebuilt per K-chunk from LDS (short live range -> no
// scratch spills) and reused across 4 column-tile WMMAs.
// ---------------------------------------------------------------------------
__global__ void knn_topk_wmma_kernel(const float* __restrict__ f, int N, int* __restrict__ idxOut) {
  const int b = blockIdx.y;
  const int rowBase = blockIdx.x * 128;
  const float* fb = f + (size_t)b * 64 * N;
  const int tid = threadIdx.x, lane = tid & 31, wave = tid >> 5;
  const int half = lane >> 4, nl = lane & 15;

  __shared__ _Float16 ldsA[128 * 72];   // [row][channel]
  __shared__ _Float16 ldsB[64 * 72];    // [col][channel]
  __shared__ float ldsXX[64];
  __shared__ float ldsPD[128 * 65];     // padded stride 65

  // stage row block once (coalesced global reads over rows)
  for (int t = tid; t < 128 * 64; t += 256) {
    int c = t >> 7, j = t & 127;
    ldsA[j * 72 + c] = (_Float16)fb[(size_t)c * N + rowBase + j];
  }
  __syncthreads();

  float xxr = 0.f;
  if (tid < 128) {
#pragma unroll
    for (int c = 0; c < 64; ++c) { float v = (float)ldsA[tid * 72 + c]; xxr += v * v; }
  }

  float val[KNNK]; int ind[KNNK];
#pragma unroll
  for (int k = 0; k < KNNK; ++k) { val[k] = -3.4e38f; ind[k] = 0; }

  for (int m0 = 0; m0 < N; m0 += 64) {
    __syncthreads();
    for (int t = tid; t < 64 * 64; t += 256) {
      int c = t >> 6, j = t & 63;
      ldsB[j * 72 + c] = (_Float16)fb[(size_t)c * N + m0 + j];
    }
    if (m0 + 64 < N)
      __builtin_prefetch(&fb[(size_t)(tid & 63) * N + m0 + 64], 0, 1);
    __syncthreads();
    if (tid < 64) {
      float s = 0.f;
#pragma unroll
      for (int c = 0; c < 64; ++c) { float v = (float)ldsB[tid * 72 + c]; s += v * v; }
      ldsXX[tid] = s;
    }

    v8f acc[4];
#pragma unroll
    for (int jt = 0; jt < 4; ++jt)
#pragma unroll
      for (int i = 0; i < 8; ++i) acc[jt][i] = 0.f;

#pragma unroll
    for (int kc = 0; kc < 64; kc += 32) {
      v8h a0 = *(const v8h*)&ldsA[(wave * 16 + nl) * 72 + kc + half * 16];
      v8h a1 = *(const v8h*)&ldsA[(wave * 16 + nl) * 72 + kc + half * 16 + 8];
      v16h A = __builtin_shufflevector(a0, a1, 0, 1, 2, 3, 4, 5, 6, 7,
                                               8, 9, 10, 11, 12, 13, 14, 15);
#pragma unroll
      for (int jt = 0; jt < 4; ++jt) {
        v16h Bv = *(const v16h*)&ldsB[(jt * 16 + nl) * 72 + kc + half * 16];
        acc[jt] = __builtin_amdgcn_wmma_f32_16x16x32_f16(
            false, A, false, Bv, (short)0, acc[jt], false, false);
      }
    }
#pragma unroll
    for (int jt = 0; jt < 4; ++jt)
#pragma unroll
      for (int i = 0; i < 8; ++i) {
        int sl = i + half * 8;
        ldsPD[(wave * 16 + sl) * 65 + jt * 16 + nl] = acc[jt][i];
      }
    __syncthreads();
    if (tid < 128) {
      for (int j = 0; j < 64; ++j) {
        float pd = 2.f * ldsPD[tid * 65 + j] - xxr - ldsXX[j];
        if (pd > val[KNNK - 1]) {
          float v = pd; int id = m0 + j;
#pragma unroll
          for (int k = 0; k < KNNK; ++k) {
            if (v > val[k]) { float tv = val[k]; int ti = ind[k]; val[k] = v; ind[k] = id; v = tv; id = ti; }
          }
        }
      }
    }
  }
  if (tid < 128) {
    int r = rowBase + tid;
#pragma unroll
    for (int k = 0; k < KNNK; ++k)
      idxOut[((size_t)b * N + r) * KNNK + k] = ind[k];
  }
}

// ---------------------------------------------------------------------------
// WMMA GEMM over "sites":  Y[b][o][site] = sum_c W[o][c] * E[c][site]
// MODE 0: E = edge features (gather via idx)      Kc = 2*C
// MODE 1: E = lrelu(instnorm(Y-in))  (in-place)   Kc = 64
// MODE 2: E = concat(x1,x2,x3)                    Kc = 192
// ---------------------------------------------------------------------------
template<int MODE>
__global__ void gemm_sites_kernel(
    const float* __restrict__ W,
    const float* __restrict__ F0, const float* __restrict__ F1, const float* __restrict__ F2,
    const int* __restrict__ idx,
    const float* __restrict__ stats,
    float* __restrict__ Y,
    float* __restrict__ partial,
    int Kc, int O, int N, int S, int wgPerBatch)
{
  __shared__ _Float16 ldsE[4 * 16 * 200];   // [wave][site][channel], stride KcPad+8
  __shared__ _Float16 ldsW[64 * 200];       // [o][channel], stride KcPad+8
  __shared__ float ldsS[4][64];
  __shared__ float ldsQ[4][64];

  const int tid  = threadIdx.x;
  const int lane = tid & 31;
  const int wave = tid >> 5;
  const int half = lane >> 4;
  const int nl   = lane & 15;
  const int b    = blockIdx.y;
  const int wg   = blockIdx.x;
  const int KcPad = (Kc + 31) & ~31;
  const int EStride = KcPad + 8;
  const int waveSite = wg * 64 + wave * 16;

  _Float16* eT = ldsE + wave * 16 * EStride;

  const int Chalf = Kc >> 1;  // MODE 0
  {
    const int sl = nl;
    const int gsite = waveSite + sl;
    int nn = 0, jn = 0;
    if (MODE == 0) {
      nn = gsite / KNNK;
      int kk = gsite - nn * KNNK;
      jn = idx[((size_t)b * N + nn) * KNNK + kk];
    }
    for (int c = half; c < KcPad; c += 2) {
      float v = 0.f;
      if (MODE == 0) {
        if (c < Chalf)
          v = F0[((size_t)b * Chalf + c) * N + jn] - F0[((size_t)b * Chalf + c) * N + nn];
        else if (c < Kc)
          v = F0[((size_t)b * Chalf + (c - Chalf)) * N + nn];
      } else if (MODE == 1) {
        if (c < Kc) {
          float raw  = F0[((size_t)b * 64 + c) * (size_t)S + gsite];
          float mean = stats[((size_t)b * 64 + c) * 2 + 0];
          float rstd = stats[((size_t)b * 64 + c) * 2 + 1];
          float z = (raw - mean) * rstd;
          v = z > 0.f ? z : 0.2f * z;
        }
      } else {
        if (c < 64)       v = F0[((size_t)b * 64 + c) * (size_t)N + gsite];
        else if (c < 128) v = F1[((size_t)b * 64 + (c - 64)) * (size_t)N + gsite];
        else if (c < 192) v = F2[((size_t)b * 64 + (c - 128)) * (size_t)N + gsite];
      }
      eT[sl * EStride + c] = (_Float16)v;
    }
  }

  for (int obase = 0; obase < O; obase += 64) {
    __syncthreads();
    for (int task = tid; task < 64 * (KcPad >> 5); task += 128) {
      int ol = task & 63;
      int cg = task >> 6;
#pragma unroll 8
      for (int cc = 0; cc < 32; ++cc) {
        int c = (cg << 5) + cc;
        float wv = (c < Kc) ? W[(size_t)(obase + ol) * Kc + c] : 0.f;
        ldsW[ol * EStride + c] = (_Float16)wv;
      }
    }
    __syncthreads();

    v8f acc[4];
#pragma unroll
    for (int t = 0; t < 4; ++t)
#pragma unroll
      for (int i = 0; i < 8; ++i) acc[t][i] = 0.f;

    for (int kc = 0; kc < KcPad; kc += 32) {
      v8h a0 = *(const v8h*)&eT[nl * EStride + kc + half * 16];
      v8h a1 = *(const v8h*)&eT[nl * EStride + kc + half * 16 + 8];
      v16h A = __builtin_shufflevector(a0, a1, 0, 1, 2, 3, 4, 5, 6, 7,
                                               8, 9, 10, 11, 12, 13, 14, 15);
#pragma unroll
      for (int t = 0; t < 4; ++t) {
        v16h Bv = *(const v16h*)&ldsW[(t * 16 + nl) * EStride + kc + half * 16];
        acc[t] = __builtin_amdgcn_wmma_f32_16x16x32_f16(
            false, A, false, Bv, (short)0, acc[t], false, false);
      }
    }

#pragma unroll
    for (int t = 0; t < 4; ++t) {
      float s = 0.f, q = 0.f;
#pragma unroll
      for (int i = 0; i < 8; ++i) {
        int sl = i + half * 8;
        int o  = obase + t * 16 + nl;
        float v = acc[t][i];
        Y[((size_t)b * O + o) * (size_t)S + (waveSite + sl)] = v;
        s += v; q += v * v;
      }
      s += __shfl_xor(s, 16);
      q += __shfl_xor(q, 16);
      if (half == 0) { ldsS[wave][t * 16 + nl] = s; ldsQ[wave][t * 16 + nl] = q; }
    }
    __syncthreads();
    if (tid < 64) {
      float s = ldsS[0][tid] + ldsS[1][tid] + ldsS[2][tid] + ldsS[3][tid];
      float q = ldsQ[0][tid] + ldsQ[1][tid] + ldsQ[2][tid] + ldsQ[3][tid];
      size_t pbase = (((size_t)b * wgPerBatch + wg) * O + obase + tid) * 2;
      partial[pbase] = s; partial[pbase + 1] = q;
    }
  }
}

__global__ void finalize_stats_kernel(const float* __restrict__ partial, float* __restrict__ stats,
                                      int nWG, int O, float invS, int total) {
  int t = blockIdx.x * 256 + threadIdx.x;
  if (t >= total) return;
  int b = t / O, o = t - b * O;
  float s = 0.f, q = 0.f;
  for (int w = 0; w < nWG; ++w) {
    size_t pbase = (((size_t)b * nWG + w) * O + o) * 2;
    s += partial[pbase]; q += partial[pbase + 1];
  }
  float mean = s * invS;
  float var  = q * invS - mean * mean;
  stats[(size_t)t * 2 + 0] = mean;
  stats[(size_t)t * 2 + 1] = rsqrtf(var + 1e-5f);
}

__global__ void maxk_kernel(const float* __restrict__ Y, const float* __restrict__ stats,
                            float* __restrict__ feat, int N, int total) {
  int t = blockIdx.x * 256 + threadIdx.x;
  if (t >= total) return;
  int n = t % N;
  int bo = t / N;
  const float* yp = Y + (size_t)bo * N * KNNK + (size_t)n * KNNK;
  float m = yp[0];
#pragma unroll
  for (int k = 1; k < KNNK; ++k) m = fmaxf(m, yp[k]);
  float mean = stats[(size_t)bo * 2], rstd = stats[(size_t)bo * 2 + 1];
  float z = (m - mean) * rstd;
  feat[(size_t)bo * N + n] = z > 0.f ? z : 0.2f * z;
}

__global__ void pool_kernel(const float* __restrict__ Y, const float* __restrict__ stats,
                            float* __restrict__ pool, int N) {
  int bo = blockIdx.x;
  const float* yp = Y + (size_t)bo * N;
  float m = -3.4e38f;
  for (int n = threadIdx.x; n < N; n += 256) m = fmaxf(m, yp[n]);
  __shared__ float red[256];
  red[threadIdx.x] = m;
  __syncthreads();
  for (int s = 128; s > 0; s >>= 1) {
    if (threadIdx.x < s) red[threadIdx.x] = fmaxf(red[threadIdx.x], red[threadIdx.x + s]);
    __syncthreads();
  }
  if (threadIdx.x == 0) {
    float mean = stats[(size_t)bo * 2], rstd = stats[(size_t)bo * 2 + 1];
    float z = (red[0] - mean) * rstd;
    pool[bo] = z > 0.f ? z : 0.2f * z;
  }
}

__global__ void assemble_kernel(const float* __restrict__ pool, const float* __restrict__ x1,
                                const float* __restrict__ x2, const float* __restrict__ x3,
                                float* __restrict__ out, int N, size_t total) {
  size_t t = (size_t)blockIdx.x * 256 + threadIdx.x;
  if (t >= total) return;
  int n = (int)(t % N);
  size_t r = t / N;
  int ch = (int)(r % 1216);
  int b  = (int)(r / 1216);
  float v;
  if (ch < 1024)      v = pool[(size_t)b * 1024 + ch];
  else if (ch < 1088) v = x1[((size_t)b * 64 + (ch - 1024)) * N + n];
  else if (ch < 1152) v = x2[((size_t)b * 64 + (ch - 1088)) * N + n];
  else                v = x3[((size_t)b * 64 + (ch - 1152)) * N + n];
  out[t] = v;
}

extern "C" void kernel_launch(void* const* d_in, const int* in_sizes, int n_in,
                              void* d_out, int out_size, void* d_ws, size_t ws_size,
                              hipStream_t stream) {
  (void)in_sizes; (void)n_in; (void)out_size; (void)ws_size;
  const float* x   = (const float*)d_in[0];
  const float* W0a = (const float*)d_in[1];
  const float* W0b = (const float*)d_in[2];
  const float* W1a = (const float*)d_in[3];
  const float* W1b = (const float*)d_in[4];
  const float* W2a = (const float*)d_in[5];
  const float* W2b = (const float*)d_in[6];
  const float* Wsm = (const float*)d_in[7];
  float* out = (float*)d_out;

  const int B = 8, N = 4096;
  const int Sconv = N * KNNK;      // 81920 sites per batch
  const int WPBc  = Sconv / 64;    // 1280 WGs per batch (conv GEMMs)
  const int WPBf  = N / 64;        // 64 WGs per batch (final GEMM)

  char* ws = (char*)d_ws;
  size_t p = 0;
  auto take = [&](size_t bytes) { char* r = ws + p; p += (bytes + 255) & ~(size_t)255; return r; };
  int*   idxB  = (int*)  take((size_t)B * N * KNNK * 4);            // 2.6 MB
  float* x1    = (float*)take((size_t)B * 64 * N * 4);              // 8.4 MB
  float* x2    = (float*)take((size_t)B * 64 * N * 4);
  float* x3    = (float*)take((size_t)B * 64 * N * 4);
  float* ybuf  = (float*)take((size_t)B * 64 * Sconv * 4);          // 168 MB (reused as 8x1024x4096)
  float* part  = (float*)take((size_t)B * WPBc * 64 * 2 * 4);       // 5.2 MB (covers final 4.2 MB)
  float* stats = (float*)take((size_t)B * 1024 * 2 * 4);
  float* pool  = (float*)take((size_t)B * 1024 * 4);

  dim3 blk128(128), blk256(256);
  dim3 gKnn3(N / 256, B);
  dim3 gKnnW(N / 128, B);
  dim3 gConv(WPBc, B);
  dim3 gFin((B * 64 + 255) / 256);
  dim3 gMax((unsigned)(((size_t)B * 64 * N + 255) / 256));

  const float* feats[3] = { x, x1, x2 };
  float* outsF[3]       = { x1, x2, x3 };
  const float* Was[3]   = { W0a, W1a, W2a };
  const float* Wbs[3]   = { W0b, W1b, W2b };
  const int Kcs[3]      = { 6, 128, 128 };

  for (int L = 0; L < 3; ++L) {
    if (L == 0)
      knn_topk_kernel<3><<<gKnn3, blk256, 0, stream>>>(feats[L], N, idxB);
    else
      knn_topk_wmma_kernel<<<gKnnW, blk256, 0, stream>>>(feats[L], N, idxB);

    gemm_sites_kernel<0><<<gConv, blk128, 0, stream>>>(
        Was[L], feats[L], nullptr, nullptr, idxB, nullptr,
        ybuf, part, Kcs[L], 64, N, Sconv, WPBc);
    finalize_stats_kernel<<<gFin, blk256, 0, stream>>>(part, stats, WPBc, 64, 1.f / Sconv, B * 64);

    gemm_sites_kernel<1><<<gConv, blk128, 0, stream>>>(
        Wbs[L], ybuf, nullptr, nullptr, nullptr, stats,
        ybuf, part, 64, 64, N, Sconv, WPBc);
    finalize_stats_kernel<<<gFin, blk256, 0, stream>>>(part, stats, WPBc, 64, 1.f / Sconv, B * 64);

    maxk_kernel<<<gMax, blk256, 0, stream>>>(ybuf, stats, outsF[L], N, B * 64 * N);
  }

  dim3 gFinal(WPBf, B);
  gemm_sites_kernel<2><<<gFinal, blk128, 0, stream>>>(
      Wsm, x1, x2, x3, nullptr, nullptr,
      ybuf, part, 192, 1024, N, N, WPBf);
  dim3 gFin2((B * 1024 + 255) / 256);
  finalize_stats_kernel<<<gFin2, blk256, 0, stream>>>(part, stats, WPBf, 1024, 1.f / N, B * 1024);
  pool_kernel<<<dim3(B * 1024), blk256, 0, stream>>>(ybuf, stats, pool, N);

  size_t totalOut = (size_t)B * 1216 * N;
  assemble_kernel<<<dim3((unsigned)((totalOut + 255) / 256)), blk256, 0, stream>>>(
      pool, x1, x2, x3, out, N, totalOut);
}
